// TetAutoEncoder_80848464380359
// MI455X (gfx1250) — matrix operand order repaired and checked by reading, not compiled, not run on hardware
//
#include <hip/hip_runtime.h>

typedef _Float16 f16;
typedef __attribute__((ext_vector_type(16))) _Float16 v16h;
typedef __attribute__((ext_vector_type(8)))  _Float16 v8h;
typedef __attribute__((ext_vector_type(8)))  float    v8f;
typedef __attribute__((ext_vector_type(4)))  unsigned int v4u;
typedef __attribute__((ext_vector_type(8)))  int v8i;
typedef __attribute__((ext_vector_type(4)))  int v4i;

#define CONV_ROWS    128
#define CONV_THREADS 256

// ---------------------------------------------------------------------------
// Weight prep: W (Cout x K fp32) -> Wc (Npad x Kpad f16, zero padded),
// column n holds Kpad contiguous halfs so B-fragments are 16B vector loads.
// ---------------------------------------------------------------------------
__global__ void wprep_kernel(const float* __restrict__ W, f16* __restrict__ wc,
                             int Cout, int K, int Kpad, int total) {
  int e = blockIdx.x * blockDim.x + threadIdx.x;
  if (e >= total) return;
  int n = e / Kpad;
  int k = e - n * Kpad;
  f16 v = (f16)0.0f;
  if (n < Cout && k < K) v = (f16)W[(size_t)n * K + k];
  wc[e] = v;
}

// ---------------------------------------------------------------------------
// Issue a TDM load of the weight tile (Npad x Kpad f16, fully contiguous)
// from global `wc` into LDS at byte offset `ldsOff`. Wave-level op.
// ---------------------------------------------------------------------------
__device__ __forceinline__ void tdm_load_weights(const f16* wc, unsigned ldsOff,
                                                 int Kpad, int Npad) {
  unsigned long long ga = (unsigned long long)(size_t)wc;
  unsigned stride01 = (unsigned)(Kpad * Npad);   // dim1 stride (elements), <2^16 here

  v4u g0;
  g0.x = 1u;                                      // count=1, user mode, no gather
  g0.y = ldsOff;                                  // lds_addr (bytes)
  g0.z = (unsigned)(ga & 0xFFFFFFFFull);          // global_addr[31:0]
  g0.w = (unsigned)((ga >> 32) & 0x1FFFFFFull)    // global_addr[56:32]
         | (2u << 30);                            // type = 2 ("image")

  v8i g1;
  g1[0] = 0x00010000;                             // wg_mask=0, data_size=1 (2B)
  g1[1] = (Kpad & 0xFFFF) << 16;                  // tensor_dim0[15:0]
  g1[2] = (Npad & 0xFFFF) << 16;                  // dim0[31:16]=0, tensor_dim1[15:0]
  g1[3] = (Kpad & 0xFFFF) << 16;                  // dim1[31:16]=0, tile_dim0
  g1[4] = (Npad & 0xFFFF);                        // tile_dim1, tile_dim2=0
  g1[5] = Kpad;                                   // tensor_dim0_stride[31:0]
  g1[6] = (int)((stride01 & 0xFFFFu) << 16);      // stride0[47:32]=0, stride1[15:0]
  g1[7] = (int)(stride01 >> 16);                  // stride1[47:16]

  v4i gz = {0, 0, 0, 0};
#if defined(__clang_major__) && (__clang_major__ >= 23)
  v8i gz8 = {0, 0, 0, 0, 0, 0, 0, 0};
  __builtin_amdgcn_tensor_load_to_lds(g0, g1, gz, gz, gz8, 0);
#else
  __builtin_amdgcn_tensor_load_to_lds(g0, g1, gz, gz, 0);
#endif
}

// ---------------------------------------------------------------------------
// tet_conv: gather 5 neighbors into LDS (f16), TDM weight tile into LDS,
// WMMA f32<-f16 16x16x32 GEMM.
// ---------------------------------------------------------------------------
__global__ __launch_bounds__(CONV_THREADS)
void tet_conv_wmma_kernel(const float* __restrict__ hin,
                          const int*   __restrict__ idx,
                          const f16*   __restrict__ wc,    // Npad x Kpad
                          const float* __restrict__ bias,
                          float*       __restrict__ hout,
                          int N, int Cin, int Cout, int Kpad, int Npad,
                          int totalRows) {
  extern __shared__ f16 As[];   // [CONV_ROWS*Kpad] A-tile | [Npad*Kpad] W-tile
  const int tid     = threadIdx.x;
  const int rowBase = blockIdx.x * CONV_ROWS;
  const int K       = 5 * Cin;

  // ---- async DMA the weight tile into LDS (wave 0 issues, all waves barrier)
  if (tid < 32) {
    unsigned ldsOff = (unsigned)(size_t)As + (unsigned)(CONV_ROWS * Kpad) * 2u;
    tdm_load_weights(wc, ldsOff, Kpad, Npad);
  }

  // ---- stage gathered A tile into LDS (f16), 2 elements per iteration ----
  if (Cin > 1) {
    const int halfPairs = Kpad >> 1;
    for (int e = tid; e < CONV_ROWS * halfPairs; e += CONV_THREADS) {
      int r  = e / halfPairs;
      int kp = e - r * halfPairs;
      int k  = kp << 1;
      int row = rowBase + r;
      unsigned packed = 0u;
      if (row < totalRows && k < K) {
        int b = row / N;
        int n = row - b * N;
        int p = k / Cin;                 // Cin even, k even -> k,k+1 same p
        int c = k - p * Cin;
        int j = idx[n * 5 + p];
        if ((unsigned)j < (unsigned)N) {
          const float* src = &hin[((size_t)b * N + j) * Cin + c];
          float2 fv = *(const float2*)src;
          union { f16 h[2]; unsigned u; } pk;
          pk.h[0] = (f16)fv.x;
          pk.h[1] = (f16)fv.y;
          packed = pk.u;
        }
      }
      *(unsigned*)&As[r * Kpad + k] = packed;
    }
  } else {
    for (int e = tid; e < CONV_ROWS * Kpad; e += CONV_THREADS) {
      int r = e / Kpad;
      int k = e - r * Kpad;
      int row = rowBase + r;
      f16 v = (f16)0.0f;
      if (row < totalRows && k < K) {
        int b = row / N;
        int n = row - b * N;
        int j = idx[n * 5 + k];          // Cin==1: p==k, c==0
        if ((unsigned)j < (unsigned)N)
          v = (f16)hin[(size_t)b * N + j];
      }
      As[r * Kpad + k] = v;
    }
  }

  if (tid < 32) __builtin_amdgcn_s_wait_tensorcnt(0);
  __syncthreads();

  const int wave  = tid >> 5;
  const int lane  = tid & 31;
  const int mlane = lane & 15;
  const int r0    = wave * 16;
  const int aHalf = (lane < 16) ? 0 : 8;    // A: 16-bit 16x32 layout
  const int bHalf = (lane < 16) ? 0 : 8;    // B: same 16-bit layout as A

  const f16* arow   = As + (size_t)(r0 + mlane) * Kpad + aHalf;
  const f16* wsBase = As + (size_t)CONV_ROWS * Kpad;   // W-tile in LDS

  const int nTiles = Npad >> 4;
  for (int tn = 0; tn < nTiles; ++tn) {
    const int ncol = tn * 16 + mlane;
    const f16* wcol = wsBase + (size_t)ncol * Kpad + bHalf;

    v8f acc = {};
    for (int kc = 0; kc < Kpad; kc += 32) {
      v8h alo = *(const v8h*)(arow + kc);
      v8h ahi = *(const v8h*)(arow + kc + 16);
      v16h a  = __builtin_shufflevector(alo, ahi, 0,1,2,3,4,5,6,7,8,9,10,11,12,13,14,15);
      v8h blo = *(const v8h*)(wcol + kc);
      v8h bhi = *(const v8h*)(wcol + kc + 16);
      v16h bf = __builtin_shufflevector(blo, bhi, 0,1,2,3,4,5,6,7,8,9,10,11,12,13,14,15);
      acc = __builtin_amdgcn_wmma_f32_16x16x32_f16(false, a, false, bf,
                                                   (short)0, acc, false, false);
    }

    if (ncol < Cout) {
      float bv = bias[ncol];
      const int mOff = (lane < 16) ? 0 : 8;
#pragma unroll
      for (int j = 0; j < 8; ++j) {
        int row = rowBase + r0 + mOff + j;
        if (row < totalRows)
          hout[(size_t)row * Cout + ncol] = acc[j] + bv;
      }
    }
  }
}

// ---------------------------------------------------------------------------
// InstanceNorm stats: one block per (b,c); stats = {mu, rstd} pairs
// ---------------------------------------------------------------------------
__global__ void inorm_stats_kernel(const float* __restrict__ h,
                                   float* __restrict__ stats, int N, int C) {
  int b = blockIdx.x / C;
  int c = blockIdx.x - b * C;
  float s = 0.f, s2 = 0.f;
  for (int n = threadIdx.x; n < N; n += blockDim.x) {
    float v = h[((size_t)b * N + n) * C + c];
    s += v; s2 += v * v;
  }
  __shared__ float sh[256], sh2[256];
  sh[threadIdx.x] = s; sh2[threadIdx.x] = s2;
  __syncthreads();
  for (int off = 128; off > 0; off >>= 1) {
    if (threadIdx.x < off) {
      sh[threadIdx.x]  += sh[threadIdx.x + off];
      sh2[threadIdx.x] += sh2[threadIdx.x + off];
    }
    __syncthreads();
  }
  if (threadIdx.x == 0) {
    float mu  = sh[0] / (float)N;
    float var = sh2[0] / (float)N - mu * mu;
    stats[2 * blockIdx.x]     = mu;
    stats[2 * blockIdx.x + 1] = rsqrtf(var + 1e-5f);
  }
}

__global__ void inorm_apply_lrelu_kernel(float* __restrict__ h,
                                         const float* __restrict__ stats,
                                         int N, int C, size_t total) {
  size_t e = (size_t)blockIdx.x * blockDim.x + threadIdx.x;
  if (e >= total) return;
  size_t NC  = (size_t)N * C;
  size_t b   = e / NC;
  size_t rem = e - b * NC;
  int c = (int)(rem % (size_t)C);
  size_t si = 2 * (b * C + c);
  float v = (h[e] - stats[si]) * stats[si + 1];
  h[e] = (v < 0.f) ? 0.2f * v : v;
}

// ---------------------------------------------------------------------------
// Max pool over groups of 8 via pool_idx gather
// ---------------------------------------------------------------------------
__global__ void pool_max_kernel(const float* __restrict__ in,
                                const int* __restrict__ pidx,
                                float* __restrict__ out,
                                int Nout, int Nin, int C, int total) {
  int e = blockIdx.x * blockDim.x + threadIdx.x;
  if (e >= total) return;
  int b   = e / (Nout * C);
  int rem = e - b * (Nout * C);
  int m = rem / C;
  int c = rem - m * C;
  float mx = -3.402823e38f;
#pragma unroll
  for (int p = 0; p < 8; ++p) {
    int j = pidx[m * 8 + p];
    float v = in[((size_t)b * Nin + j) * C + c];
    mx = fmaxf(mx, v);
  }
  out[e] = mx;
}

__global__ void unpool_kernel(const float* __restrict__ in,
                              const int* __restrict__ uidx,
                              float* __restrict__ out,
                              int Nout, int Nin, int C, int total) {
  int e = blockIdx.x * blockDim.x + threadIdx.x;
  if (e >= total) return;
  int b   = e / (Nout * C);
  int rem = e - b * (Nout * C);
  int i = rem / C;
  int c = rem - i * C;
  out[e] = in[((size_t)b * Nin + uidx[i]) * C + c];
}

// ---------------------------------------------------------------------------
// FC kernels
// ---------------------------------------------------------------------------
__global__ void fc_big_kernel(const float* __restrict__ x,   // B x K
                              const float* __restrict__ W,   // O x K
                              const float* __restrict__ bias,
                              float* __restrict__ out, int K, int O,
                              int applyLrelu) {
  int b = blockIdx.x / O;
  int o = blockIdx.x - b * O;
  const float* xp = x + (size_t)b * K;
  const float* wp = W + (size_t)o * K;
  float s = 0.f;
  for (int k = threadIdx.x; k < K; k += blockDim.x) s += xp[k] * wp[k];
  __shared__ float sh[256];
  sh[threadIdx.x] = s;
  __syncthreads();
  for (int off = 128; off > 0; off >>= 1) {
    if (threadIdx.x < off) sh[threadIdx.x] += sh[threadIdx.x + off];
    __syncthreads();
  }
  if (threadIdx.x == 0) {
    float v = sh[0] + bias[o];
    if (applyLrelu) v = (v < 0.f) ? 0.2f * v : v;
    out[b * O + o] = v;
  }
}

__global__ void fc_small_kernel(const float* __restrict__ x,  // B x K
                                const float* __restrict__ W,  // O x K
                                const float* __restrict__ bias,
                                float* __restrict__ out1,
                                float* __restrict__ out2,
                                int Bsz, int K, int O) {
  int e = blockIdx.x * blockDim.x + threadIdx.x;
  if (e >= Bsz * O) return;
  int b = e / O;
  int o = e - b * O;
  float s = bias[o];
  for (int k = 0; k < K; ++k) s += x[b * K + k] * W[(size_t)o * K + k];
  out1[e] = s;
  if (out2) out2[e] = s;
}

__global__ void fc_dec_kernel(const float* __restrict__ x,   // B x 64
                              const float* __restrict__ W,   // O x 64
                              const float* __restrict__ bias,
                              float* __restrict__ out, int Bsz, int O) {
  int e = blockIdx.x * blockDim.x + threadIdx.x;
  if (e >= Bsz * O) return;
  int b = e / O;
  int i = e - b * O;
  float s = bias[i];
#pragma unroll
  for (int c = 0; c < 64; ++c) s += x[b * 64 + c] * W[(size_t)i * 64 + c];
  out[e] = s;
}

// ---------------------------------------------------------------------------
// Host orchestration
// ---------------------------------------------------------------------------
extern "C" void kernel_launch(void* const* d_in, const int* in_sizes, int n_in,
                              void* d_out, int out_size, void* d_ws, size_t ws_size,
                              hipStream_t stream) {
  (void)in_sizes; (void)n_in; (void)out_size; (void)ws_size;
  const int Bsz = 8;
  const int N2 = 65536, N1 = 8192, N0 = 1024;

  // allow up to 160KB dynamic LDS for the conv kernel (non-stream API)
  (void)hipFuncSetAttribute((const void*)tet_conv_wmma_kernel,
                            hipFuncAttributeMaxDynamicSharedMemorySize, 160 * 1024);

  // --- input mapping ---
  const float* x = (const float*)d_in[0];
  auto encW = [&](int l) { return (const float*)d_in[1 + 2 * l]; };
  auto encB = [&](int l) { return (const float*)d_in[2 + 2 * l]; };
  const float* encfcW  = (const float*)d_in[25];
  const float* encfcB  = (const float*)d_in[26];
  const float* meanW   = (const float*)d_in[27];
  const float* meanB   = (const float*)d_in[28];
  const float* logvW   = (const float*)d_in[29];
  const float* logvB   = (const float*)d_in[30];
  const float* decfcW  = (const float*)d_in[31];
  const float* decfcB  = (const float*)d_in[32];
  auto decW = [&](int l) { return (const float*)d_in[33 + 2 * l]; };
  auto decB = [&](int l) { return (const float*)d_in[34 + 2 * l]; };
  const int* idx0 = (const int*)d_in[57];   // N=1024
  const int* idx1 = (const int*)d_in[58];   // N=8192
  const int* idx2 = (const int*)d_in[59];   // N=65536
  const int* pool1 = (const int*)d_in[60];  // len 8192  (8192->1024)
  const int* pool2 = (const int*)d_in[61];  // len 65536 (65536->8192)
  const int* unp1  = (const int*)d_in[62];  // len 8192
  const int* unp2  = (const int*)d_in[63];  // len 65536

  // --- workspace layout ---
  const size_t S = (size_t)Bsz * N2 * 16;        // 8,388,608 floats
  float* bufA    = (float*)d_ws;
  float* bufB    = bufA + S;
  float* statbuf = bufB + S;                     // 2*8*64
  float* zbuf    = statbuf + 1024;               // 512
  float* meanbuf = zbuf + 512;                   // 512
  f16*   wcbuf   = (f16*)(meanbuf + 512);        // <= 64*320 halfs

  float* out_h    = (float*)d_out;
  float* out_mean = out_h + (size_t)Bsz * N2;
  float* out_logv = out_mean + Bsz * 64;

  auto conv = [&](const float* hin, float* hout, const int* idx,
                  const float* W, const float* bias, int N, int Cin, int Cout) {
    int K    = 5 * Cin;
    int Kpad = (K + 31) & ~31;
    int Npad = (Cout + 15) & ~15;
    int wtot = Npad * Kpad;
    wprep_kernel<<<(wtot + 255) / 256, 256, 0, stream>>>(W, wcbuf, Cout, K, Kpad, wtot);
    int totalRows = Bsz * N;
    int blocks = (totalRows + CONV_ROWS - 1) / CONV_ROWS;
    size_t ldsBytes = (size_t)(CONV_ROWS + Npad) * Kpad * sizeof(f16);
    tet_conv_wmma_kernel<<<blocks, CONV_THREADS, ldsBytes, stream>>>(
        hin, idx, wcbuf, bias, hout, N, Cin, Cout, Kpad, Npad, totalRows);
  };

  auto inlrelu = [&](float* h, int N, int C) {
    inorm_stats_kernel<<<Bsz * C, 256, 0, stream>>>(h, statbuf, N, C);
    size_t total = (size_t)Bsz * N * C;
    inorm_apply_lrelu_kernel<<<(int)((total + 255) / 256), 256, 0, stream>>>(
        h, statbuf, N, C, total);
  };

  float* A = bufA;
  float* Bb = bufB;
  auto swap = [&]() { float* t = A; A = Bb; Bb = t; };

  // ===================== Encoder =====================
  // level 0: N=65536
  conv(x, A, idx2, encW(0), encB(0), N2, 1, 16);
  for (int k = 1; k < 4; ++k) {
    inlrelu(A, N2, 16);
    conv(A, Bb, idx2, encW(k), encB(k), N2, 16, 16);
    swap();
  }
  { int total = Bsz * N1 * 16;
    pool_max_kernel<<<(total + 255) / 256, 256, 0, stream>>>(A, pool2, Bb, N1, N2, 16, total);
    swap(); }
  inlrelu(A, N1, 16);

  // level 1: N=8192
  conv(A, Bb, idx1, encW(4), encB(4), N1, 16, 32); swap();
  for (int k = 5; k < 8; ++k) {
    inlrelu(A, N1, 32);
    conv(A, Bb, idx1, encW(k), encB(k), N1, 32, 32);
    swap();
  }
  { int total = Bsz * N0 * 32;
    pool_max_kernel<<<(total + 255) / 256, 256, 0, stream>>>(A, pool1, Bb, N0, N1, 32, total);
    swap(); }
  inlrelu(A, N0, 32);

  // level 2: N=1024
  conv(A, Bb, idx0, encW(8), encB(8), N0, 32, 64); swap();
  for (int k = 9; k < 12; ++k) {
    inlrelu(A, N0, 64);
    conv(A, Bb, idx0, encW(k), encB(k), N0, 64, 64);
    swap();
  }
  inlrelu(A, N0, 64);

  // ===================== Bottleneck =====================
  fc_big_kernel<<<Bsz * 64, 256, 0, stream>>>(A, encfcW, encfcB, zbuf, 65536, 64, 1);
  fc_small_kernel<<<(Bsz * 64 + 255) / 256, 256, 0, stream>>>(
      zbuf, meanW, meanB, out_mean, meanbuf, Bsz, 64, 64);
  fc_small_kernel<<<(Bsz * 64 + 255) / 256, 256, 0, stream>>>(
      zbuf, logvW, logvB, out_logv, nullptr, Bsz, 64, 64);
  { int total = Bsz * 65536;
    fc_dec_kernel<<<(total + 255) / 256, 256, 0, stream>>>(
        meanbuf, decfcW, decfcB, A, Bsz, 65536); }
  inlrelu(A, N0, 64);

  // ===================== Decoder =====================
  // level 0: N=1024, 64->64,64,64,32
  conv(A, Bb, idx0, decW(0), decB(0), N0, 64, 64); swap();
  for (int k = 1; k < 3; ++k) {
    inlrelu(A, N0, 64);
    conv(A, Bb, idx0, decW(k), decB(k), N0, 64, 64);
    swap();
  }
  inlrelu(A, N0, 64);
  conv(A, Bb, idx0, decW(3), decB(3), N0, 64, 32); swap();
  inlrelu(A, N0, 32);

  // level 1: unpool 1024->8192, 32->32,32,32,16
  { int total = Bsz * N1 * 32;
    unpool_kernel<<<(total + 255) / 256, 256, 0, stream>>>(A, unp1, Bb, N1, N0, 32, total);
    swap(); }
  conv(A, Bb, idx1, decW(4), decB(4), N1, 32, 32); swap();
  for (int k = 5; k < 7; ++k) {
    inlrelu(A, N1, 32);
    conv(A, Bb, idx1, decW(k), decB(k), N1, 32, 32);
    swap();
  }
  inlrelu(A, N1, 32);
  conv(A, Bb, idx1, decW(7), decB(7), N1, 32, 16); swap();
  inlrelu(A, N1, 16);

  // level 2: unpool 8192->65536, 16->16,16,16,1 (no trailing norm)
  { int total = Bsz * N2 * 16;
    unpool_kernel<<<(total + 255) / 256, 256, 0, stream>>>(A, unp2, Bb, N2, N1, 16, total);
    swap(); }
  conv(A, Bb, idx2, decW(8), decB(8), N2, 16, 16); swap();
  for (int k = 9; k < 11; ++k) {
    inlrelu(A, N2, 16);
    conv(A, Bb, idx2, decW(k), decB(k), N2, 16, 16);
    swap();
  }
  inlrelu(A, N2, 16);
  conv(A, out_h, idx2, decW(11), decB(11), N2, 16, 1);   // final h -> d_out
}
